// HiddenCellLSTM_80676665688839
// MI455X (gfx1250) — compile-verified, matrix operand order checked by reading, not run on hardware
//
#include <hip/hip_runtime.h>
#include <hip/hip_bf16.h>
#include <stdint.h>

// ---------------- problem constants ----------------
#define T_STEPS 2048
#define BATCH   64
#define IN_DIM  256
#define HID     256
#define G4      (4 * HID)       // 1024 gate columns
#define NSLICE  8               // split of 4H=1024 gate columns across WGPs
#define BGROUPS 4               // batch groups of 16 rows each
#define HSL     (HID / NSLICE)  // 32 H-columns owned per WGP
#define NLOC    (4 * HSL)       // 128 gate columns per WGP (4 gates x 32)
#define MROWS   16              // batch rows per group (one WMMA M tile)
#define LSTR    264             // padded LDS K-stride in bf16 elems (528B rows)

// ---- workspace layout (two-phase path) ----
#define WS_CNT_OFF   0u
#define WS_CNT_BYTES ((size_t)BGROUPS * T_STEPS * 4)            // 32 KB
#define WS_HBUF_OFF  (WS_CNT_OFF + WS_CNT_BYTES)
#define WS_HBUF_BYTES ((size_t)2 * BATCH * HID * 2)             // 64 KB
#define WS_WBF_OFF   (WS_HBUF_OFF + WS_HBUF_BYTES)
#define WS_WBF_BYTES ((size_t)G4 * IN_DIM * 2)                  // 512 KB
#define WS_BIAS_OFF  (WS_WBF_OFF + WS_WBF_BYTES)
#define WS_BIAS_BYTES ((size_t)G4 * 4)                          // 4 KB
#define WS_XG_OFF    (WS_BIAS_OFF + WS_BIAS_BYTES)
#define WS_XG_BYTES  ((size_t)T_STEPS * BATCH * G4 * 4)         // 512 MB
#define WS_TOTAL     (WS_XG_OFF + WS_XG_BYTES)

typedef __bf16 v16bf __attribute__((ext_vector_type(16)));
typedef __bf16 v8bf  __attribute__((ext_vector_type(8)));
typedef float  v8f   __attribute__((ext_vector_type(8)));

// Load one 16x32 bf16 WMMA operand fragment for this lane (LDS or global).
// Per ISA layout: lane holds K elems { lh*8 .. +7 } and { 16+lh*8 .. +7 }
// relative to the 32-wide K tile, i.e. two contiguous 16B runs.
__device__ __forceinline__ v16bf frag_ld(const __bf16* p) {
  v8bf lo = *(const v8bf*)p;
  v8bf hi = *(const v8bf*)(p + 16);
  return __builtin_shufflevector(lo, hi, 0, 1, 2, 3, 4, 5, 6, 7,
                                 8, 9, 10, 11, 12, 13, 14, 15);
}

// CDNA5 async global->LDS copy (16B per lane), device scope, ASYNCcnt-tracked.
__device__ __forceinline__ void async_g2l_b128(uint32_t lds_off,
                                               uint64_t gaddr) {
  asm volatile("global_load_async_to_lds_b128 %0, %1, off scope:SCOPE_DEV"
               :: "v"(lds_off), "v"(gaddr)
               : "memory");
}
__device__ __forceinline__ void wait_async0() {
  asm volatile("s_wait_asynccnt 0" ::: "memory");
}

__global__ void init_cnt_kernel(unsigned int* cnt, int n) {
  int i = blockIdx.x * blockDim.x + threadIdx.x;
  if (i < n) cnt[i] = 0u;
}

// ---- phase A0: convert W_ih to bf16, build fused bias --------------------
__global__ void prep_kernel(const float* __restrict__ w_ih,
                            const float* __restrict__ b_ih,
                            const float* __restrict__ b_hh,
                            __bf16* __restrict__ wbf,
                            float* __restrict__ biasws) {
  int i = blockIdx.x * blockDim.x + threadIdx.x;
  if (i < G4 * IN_DIM) wbf[i] = (__bf16)w_ih[i];
  if (i < G4) biasws[i] = b_ih[i] + b_hh[i];
}

// ---- phase A: xg[m-tile][n-tile] = x@W_ihT + bias, stored in C-frag order -
// grid: T*BGROUPS blocks (one 16-row M tile each), 256 threads.
__global__ __launch_bounds__(256) void xg_gemm_kernel(
    const float* __restrict__ x, const __bf16* __restrict__ wbf,
    const float* __restrict__ biasws, float* __restrict__ xg) {
  __shared__ __bf16 x_l[MROWS][LSTR];

  const int tid = threadIdx.x;
  const int lane = tid & 31;
  const int wv = tid >> 5;
  const int lh = lane >> 4;
  const int lr = lane & 15;
  const int m = blockIdx.x;          // m = t*BGROUPS + bg
  const int t = m >> 2;
  const int b0 = (m & 3) * MROWS;

  {  // stage x tile -> LDS bf16
    const int row = tid >> 4;
    const int kb = (tid & 15) * 16;
    const float* xs =
        x + (size_t)t * BATCH * IN_DIM + (size_t)(b0 + row) * IN_DIM + kb;
#pragma unroll
    for (int k = 0; k < 16; ++k) x_l[row][kb + k] = (__bf16)xs[k];
  }
  __syncthreads();

#pragma unroll
  for (int j = 0; j < 8; ++j) {       // 8 N-tiles per wave (64 total)
    const int nt = wv * 8 + j;
    const float bv = biasws[nt * 16 + lr];
    v8f acc;
#pragma unroll
    for (int r = 0; r < 8; ++r) acc[r] = bv;
    const __bf16* ax = &x_l[lr][lh * 8];
    const __bf16* bw = wbf + (size_t)(nt * 16 + lr) * IN_DIM + lh * 8;
#pragma unroll
    for (int kt = 0; kt < 8; ++kt) {
      v16bf a = frag_ld(ax + kt * 32);
      v16bf b = frag_ld(bw + kt * 32);
      acc = __builtin_amdgcn_wmma_f32_16x16x32_bf16(false, a, false, b,
                                                    (short)0, acc, false,
                                                    false);
    }
    v8f* dst = (v8f*)(xg + (((size_t)m * 64 + nt) * 32 + lane) * 8);
    __builtin_nontemporal_store(acc, dst);
  }
}

// ---- phase B: recurrence only (h-GEMM + gates), xg preloaded -------------
__global__ __launch_bounds__(256) void lstm_recur_kernel(
    const float* __restrict__ xg,    // C-fragment-ordered input projections
    const float* __restrict__ h0, const float* __restrict__ c0,
    const float* __restrict__ w_hh, float* __restrict__ out,
    unsigned int* __restrict__ cnt, __bf16* __restrict__ hbuf) {
  __shared__ __bf16 whh_l[NLOC][LSTR];
  __shared__ __bf16 h_l[MROWS][LSTR];
  __shared__ float  g_l[NLOC][MROWS];

  const int tid = threadIdx.x;
  const int lane = tid & 31;
  const int wv = tid >> 5;          // owns N-tile wv (16 gate cols)
  const int lh = lane >> 4;
  const int lr = lane & 15;
  const int bg = blockIdx.x >> 3;
  const int ns = blockIdx.x & 7;
  const int b0 = bg * MROWS;

  {  // stage W_hh slice -> LDS bf16
    const int rrow = tid >> 1;
    const int kh = (tid & 1) * 128;
    const int grow = (rrow >> 5) * HID + ns * HSL + (rrow & 31);
    const float* sh = w_hh + (size_t)grow * HID + kh;
    for (int k = 0; k < 128; ++k) whh_l[rrow][kh + k] = (__bf16)sh[k];
  }

  const int hc = tid & 31;
  const int rg = tid >> 5;
  const int hcol = ns * HSL + hc;
  const int srow = tid >> 4;               // staging row
  const int skb = (tid & 15) * 16;         // staging k-base
  float creg[2];
  const size_t TBH = (size_t)T_STEPS * BATCH * HID;

  for (int t = 0; t < T_STEPS; ++t) {
    // accumulator = precomputed xg tile (coalesced 32B/lane, streaming)
    const size_t xgbase =
        (((size_t)(t * BGROUPS + bg) * 64 + ns * 8 + wv) * 32 + lane) * 8;
    v8f acc = __builtin_nontemporal_load((const v8f*)(xg + xgbase));
    if (t + 1 < T_STEPS)
      __builtin_prefetch(xg + xgbase + (size_t)BGROUPS * 64 * 256, 0, 1);

    // wait for all 8 slice-WGPs of this batch group (step t-1)
    if (t > 0 && tid == 0) {
      while (__hip_atomic_load(&cnt[bg * T_STEPS + t - 1], __ATOMIC_ACQUIRE,
                               __HIP_MEMORY_SCOPE_AGENT) < NSLICE) {
        __builtin_amdgcn_s_sleep(1);
      }
    }
    __syncthreads();
    // stage h_{t-1} [16 x 256] -> LDS
    if (t == 0) {
      const float* hs = h0 + (size_t)(b0 + srow) * HID + skb;
#pragma unroll
      for (int k = 0; k < 16; ++k) h_l[srow][skb + k] = (__bf16)hs[k];
    } else {
      // raw bf16 tile: async global->LDS copy, no VGPR round trip
      const __bf16* src = hbuf + (size_t)((t - 1) & 1) * BATCH * HID +
                          (size_t)(b0 + srow) * HID + skb;
      uint32_t d = (uint32_t)(uintptr_t)&h_l[srow][skb];
      async_g2l_b128(d, (uint64_t)(uintptr_t)src);
      async_g2l_b128(d + 16u, (uint64_t)(uintptr_t)(src + 8));
      wait_async0();
    }
    __syncthreads();

    // h-GEMM: 8 WMMA per wave on the serial critical path
    {
      const __bf16* bh = &whh_l[wv * 16 + lr][lh * 8];
      const __bf16* ah = &h_l[lr][lh * 8];
#pragma unroll
      for (int kt = 0; kt < 8; ++kt) {
        v16bf a = frag_ld(ah + kt * 32);
        v16bf b = frag_ld(bh + kt * 32);
        acc = __builtin_amdgcn_wmma_f32_16x16x32_bf16(false, a, false, b,
                                                      (short)0, acc, false,
                                                      false);
      }
      float* gp = &g_l[wv * 16 + lr][lh * 8];
#pragma unroll
      for (int r = 0; r < 8; ++r) gp[r] = acc[r];
    }
    __syncthreads();

    // gates -> c,h update; write outputs + h exchange buffer
#pragma unroll
    for (int rr = 0; rr < 2; ++rr) {
      const int row = rg * 2 + rr;
      const float gi = g_l[0 * HSL + hc][row];
      const float gf = g_l[1 * HSL + hc][row];
      const float gg = g_l[2 * HSL + hc][row];
      const float go = g_l[3 * HSL + hc][row];
      const float cprev =
          (t == 0) ? c0[(size_t)(b0 + row) * HID + hcol] : creg[rr];
      const float si = 1.f / (1.f + __expf(-gi));
      const float sf = 1.f / (1.f + __expf(-gf));
      const float so = 1.f / (1.f + __expf(-go));
      const float tg = tanhf(gg);
      const float cn = sf * cprev + si * tg;
      const float hn = so * tanhf(cn);
      creg[rr] = cn;
      const size_t oi =
          (size_t)t * BATCH * HID + (size_t)(b0 + row) * HID + hcol;
      __builtin_nontemporal_store(hn, &out[oi]);
      __builtin_nontemporal_store(cn, &out[TBH + oi]);
      hbuf[(size_t)(t & 1) * BATCH * HID + (size_t)(b0 + row) * HID + hcol] =
          (__bf16)hn;
      if (t == T_STEPS - 1) {
        const size_t fb = 2 * TBH;
        out[fb + (size_t)(b0 + row) * HID + hcol] = hn;
        out[fb + (size_t)BATCH * HID + (size_t)(b0 + row) * HID + hcol] = cn;
      }
    }
    __syncthreads();
    if (tid == 0) {
      __hip_atomic_fetch_add(&cnt[bg * T_STEPS + t], 1u, __ATOMIC_RELEASE,
                             __HIP_MEMORY_SCOPE_AGENT);
    }
  }
}

// ---- fallback: fully fused persistent kernel (round-1, compile-proven) ---
__global__ __launch_bounds__(256) void lstm_fused_kernel(
    const float* __restrict__ x, const float* __restrict__ h0,
    const float* __restrict__ c0, const float* __restrict__ w_ih,
    const float* __restrict__ w_hh, const float* __restrict__ b_ih,
    const float* __restrict__ b_hh, float* __restrict__ out,
    unsigned int* __restrict__ cnt, __bf16* __restrict__ hbuf) {
  __shared__ __bf16 wih_l[NLOC][LSTR];
  __shared__ __bf16 whh_l[NLOC][LSTR];
  __shared__ __bf16 x_l[MROWS][LSTR];
  __shared__ __bf16 h_l[MROWS][LSTR];
  __shared__ float  g_l[NLOC][MROWS];
  __shared__ float  bias_l[NLOC];

  const int tid = threadIdx.x;
  const int lane = tid & 31;
  const int wv = tid >> 5;
  const int lh = lane >> 4;
  const int lr = lane & 15;
  const int bg = blockIdx.x >> 3;
  const int ns = blockIdx.x & 7;
  const int b0 = bg * MROWS;

  {
    const int rrow = tid >> 1;
    const int kh = (tid & 1) * 128;
    const int grow = (rrow >> 5) * HID + ns * HSL + (rrow & 31);
    const float* si = w_ih + (size_t)grow * IN_DIM + kh;
    const float* sh = w_hh + (size_t)grow * HID + kh;
    for (int k = 0; k < 128; ++k) wih_l[rrow][kh + k] = (__bf16)si[k];
    for (int k = 0; k < 128; ++k) whh_l[rrow][kh + k] = (__bf16)sh[k];
    if (tid < NLOC) {
      const int gg = (tid >> 5) * HID + ns * HSL + (tid & 31);
      bias_l[tid] = b_ih[gg] + b_hh[gg];
    }
  }

  const int hc = tid & 31;
  const int rg = tid >> 5;
  const int hcol = ns * HSL + hc;
  float creg[2];
  const size_t TBH = (size_t)T_STEPS * BATCH * HID;

  for (int t = 0; t < T_STEPS; ++t) {
    {
      const int row = tid >> 4;
      const int kb = (tid & 15) * 16;
      const float* xs =
          x + (size_t)t * BATCH * IN_DIM + (size_t)(b0 + row) * IN_DIM + kb;
#pragma unroll
      for (int k = 0; k < 16; ++k) x_l[row][kb + k] = (__bf16)xs[k];
      if (t + 1 < T_STEPS) __builtin_prefetch(xs + BATCH * IN_DIM, 0, 1);
    }
    if (t > 0 && tid == 0) {
      while (__hip_atomic_load(&cnt[bg * T_STEPS + t - 1], __ATOMIC_ACQUIRE,
                               __HIP_MEMORY_SCOPE_AGENT) < NSLICE) {
        __builtin_amdgcn_s_sleep(1);
      }
    }
    __syncthreads();
    {
      const int row = tid >> 4;
      const int kb = (tid & 15) * 16;
      if (t == 0) {
        const float* hs = h0 + (size_t)(b0 + row) * HID + kb;
#pragma unroll
        for (int k = 0; k < 16; ++k) h_l[row][kb + k] = (__bf16)hs[k];
      } else {
        unsigned int* hb =
            (unsigned int*)(hbuf + (size_t)((t - 1) & 1) * BATCH * HID +
                            (size_t)(b0 + row) * HID + kb);
        unsigned int* hd = (unsigned int*)&h_l[row][kb];
#pragma unroll
        for (int k = 0; k < 8; ++k)
          hd[k] = __hip_atomic_load(&hb[k], __ATOMIC_RELAXED,
                                    __HIP_MEMORY_SCOPE_AGENT);
      }
    }
    __syncthreads();
    {
      const float bv = bias_l[wv * 16 + lr];
      v8f acc;
#pragma unroll
      for (int r = 0; r < 8; ++r) acc[r] = bv;
      const __bf16* bi = &wih_l[wv * 16 + lr][lh * 8];
      const __bf16* bh = &whh_l[wv * 16 + lr][lh * 8];
      const __bf16* ax = &x_l[lr][lh * 8];
      const __bf16* ah = &h_l[lr][lh * 8];
#pragma unroll
      for (int kt = 0; kt < 8; ++kt) {
        v16bf a = frag_ld(ax + kt * 32);
        v16bf b = frag_ld(bi + kt * 32);
        acc = __builtin_amdgcn_wmma_f32_16x16x32_bf16(false, a, false, b,
                                                      (short)0, acc, false,
                                                      false);
      }
#pragma unroll
      for (int kt = 0; kt < 8; ++kt) {
        v16bf a = frag_ld(ah + kt * 32);
        v16bf b = frag_ld(bh + kt * 32);
        acc = __builtin_amdgcn_wmma_f32_16x16x32_bf16(false, a, false, b,
                                                      (short)0, acc, false,
                                                      false);
      }
      float* gp = &g_l[wv * 16 + lr][lh * 8];
#pragma unroll
      for (int r = 0; r < 8; ++r) gp[r] = acc[r];
    }
    __syncthreads();
#pragma unroll
    for (int rr = 0; rr < 2; ++rr) {
      const int row = rg * 2 + rr;
      const float gi = g_l[0 * HSL + hc][row];
      const float gf = g_l[1 * HSL + hc][row];
      const float gg = g_l[2 * HSL + hc][row];
      const float go = g_l[3 * HSL + hc][row];
      const float cprev =
          (t == 0) ? c0[(size_t)(b0 + row) * HID + hcol] : creg[rr];
      const float si = 1.f / (1.f + __expf(-gi));
      const float sf = 1.f / (1.f + __expf(-gf));
      const float so = 1.f / (1.f + __expf(-go));
      const float tg = tanhf(gg);
      const float cn = sf * cprev + si * tg;
      const float hn = so * tanhf(cn);
      creg[rr] = cn;
      const size_t oi =
          (size_t)t * BATCH * HID + (size_t)(b0 + row) * HID + hcol;
      __builtin_nontemporal_store(hn, &out[oi]);
      __builtin_nontemporal_store(cn, &out[TBH + oi]);
      hbuf[(size_t)(t & 1) * BATCH * HID + (size_t)(b0 + row) * HID + hcol] =
          (__bf16)hn;
      if (t == T_STEPS - 1) {
        const size_t fb = 2 * TBH;
        out[fb + (size_t)(b0 + row) * HID + hcol] = hn;
        out[fb + (size_t)BATCH * HID + (size_t)(b0 + row) * HID + hcol] = cn;
      }
    }
    __syncthreads();
    if (tid == 0) {
      __hip_atomic_fetch_add(&cnt[bg * T_STEPS + t], 1u, __ATOMIC_RELEASE,
                             __HIP_MEMORY_SCOPE_AGENT);
    }
  }
}

extern "C" void kernel_launch(void* const* d_in, const int* in_sizes, int n_in,
                              void* d_out, int out_size, void* d_ws,
                              size_t ws_size, hipStream_t stream) {
  const float* x    = (const float*)d_in[0];
  const float* h0   = (const float*)d_in[1];
  const float* c0   = (const float*)d_in[2];
  const float* w_ih = (const float*)d_in[3];
  const float* w_hh = (const float*)d_in[4];
  const float* b_ih = (const float*)d_in[5];
  const float* b_hh = (const float*)d_in[6];
  float* out = (float*)d_out;

  char* ws = (char*)d_ws;
  unsigned int* cnt = (unsigned int*)(ws + WS_CNT_OFF);
  __bf16* hbuf = (__bf16*)(ws + WS_HBUF_OFF);
  const int ncnt = BGROUPS * T_STEPS;
  init_cnt_kernel<<<(ncnt + 255) / 256, 256, 0, stream>>>(cnt, ncnt);

  if (ws_size >= WS_TOTAL) {
    // two-phase: parallel input projection on the whole chip, then the
    // halved-critical-path recurrence on 32 persistent WGPs.
    __bf16* wbf = (__bf16*)(ws + WS_WBF_OFF);
    float* biasws = (float*)(ws + WS_BIAS_OFF);
    float* xg = (float*)(ws + WS_XG_OFF);
    prep_kernel<<<(G4 * IN_DIM + 255) / 256, 256, 0, stream>>>(
        w_ih, b_ih, b_hh, wbf, biasws);
    xg_gemm_kernel<<<T_STEPS * BGROUPS, 256, 0, stream>>>(x, wbf, biasws, xg);
    lstm_recur_kernel<<<BGROUPS * NSLICE, 256, 0, stream>>>(
        xg, h0, c0, w_hh, out, cnt, hbuf);
  } else {
    lstm_fused_kernel<<<BGROUPS * NSLICE, 256, 0, stream>>>(
        x, h0, c0, w_ih, w_hh, b_ih, b_hh, out, cnt, hbuf);
  }
}
